// RV2BEVFrustumAttn_11184094839416
// MI455X (gfx1250) — matrix operand order repaired and checked by reading, not compiled, not run on hardware
//
#include <hip/hip_runtime.h>
#include <hip/hip_bf16.h>
#include <math.h>

typedef __bf16 bf16;
typedef __attribute__((ext_vector_type(16))) __bf16 v16bf;
typedef __attribute__((ext_vector_type(8)))  float  v8f;

typedef int v4i_ __attribute__((vector_size(16)));
typedef __attribute__((address_space(1))) v4i_* gptr4;
typedef __attribute__((address_space(3))) v4i_* lptr4;

#if defined(__AMDGCN__) && __has_builtin(__builtin_amdgcn_global_load_async_to_lds_b128) && __has_builtin(__builtin_amdgcn_s_wait_asynccnt)
#define USE_ASYNC_LDS 1
#else
#define USE_ASYNC_LDS 0
#endif

__device__ __forceinline__ float gelu_exact(float x) {
    return 0.5f * x * (1.0f + erff(x * 0.70710678118654752440f));
}

union FragU { uint4 u[2]; v16bf v; };

// ---------------------------------------------------------------------------
// WMMA GEMM: C[M,N] = A[M,K](bf16 row) * W[N,K](bf16 row)^T + bias, epilogues.
// EPI: 0 none, 1 exact GELU, 2 add f32 aux[M,N].
// block = 128 threads (4 waves); wave -> 16 x (16*NTILES) strip.
// B tile staged per 64-K chunk into LDS in per-lane fragment order
// (conflict-free, 16B-aligned ds_load_b128), async double-buffered when the
// gfx1250 async-to-LDS builtins are available.
// grid = (N/(16*NTILES), M/64). Requires M%64==0, N%(16*NTILES)==0, K%64==0.
// ---------------------------------------------------------------------------
template<int NTILES, int EPI>
__global__ void gemm_wmma_kernel(const bf16* __restrict__ A,
                                 const bf16* __restrict__ W,
                                 const float* __restrict__ bias,
                                 const float* __restrict__ aux,
                                 float* __restrict__ Cf,
                                 bf16*  __restrict__ Cb,
                                 int M, int N, int K)
{
    __shared__ bf16 Bsm[2][4096];            // 2 x 8KB (NTILES==4 worst case)

    const int lane = threadIdx.x & 31;
    const int wave = threadIdx.x >> 5;
    const int row0 = (blockIdx.y * 4 + wave) * 16;
    const int col0 = blockIdx.x * (16 * NTILES);

    const int m    = lane & 15;
    const int half = lane >> 4;
    const int kbA  = half * 8;               // A lane-half K base

    // ---- cooperative B staging: 16B unit u -> LDS byte offset u*16 ----
    // u = ((kstep*NTILES + t)*32 + lane)*2 + half16B
    auto stageB = [&](int kk, int bufsel) {
#pragma unroll
        for (int i = 0; i < NTILES; ++i) {
            int u     = threadIdx.x + i * 128;
            int halfu = u & 1;
            int laneu = (u >> 1) & 31;
            int rest  = u >> 6;              // kstep*NTILES + t
            int tt    = rest % NTILES;
            int kst   = rest / NTILES;
            int col   = col0 + tt * 16 + (laneu & 15);
            int krow  = kk + kst * 32 + ((laneu >> 4) << 3) + halfu * 16;
            const bf16* gsrc = W + (size_t)col * K + krow;
            bf16* ldst = &Bsm[bufsel][(size_t)u * 8];
#if USE_ASYNC_LDS
            __builtin_amdgcn_global_load_async_to_lds_b128(
                (gptr4)(void*)gsrc, (lptr4)(void*)ldst, 0, 0);
#else
            *(uint4*)ldst = *(const uint4*)gsrc;
#endif
        }
    };

    v8f acc[NTILES];
#pragma unroll
    for (int t = 0; t < NTILES; ++t) {
#pragma unroll
        for (int j = 0; j < 8; ++j) acc[t][j] = 0.0f;
    }

    const bf16* arow = A + (size_t)(row0 + m) * K;
    int buf = 0;

#if USE_ASYNC_LDS
    stageB(0, 0);
#endif

    for (int kk = 0; kk < K; kk += 64) {
        // A fragments for both k-steps of this chunk (latency overlaps staging wait)
        const bf16* ap = arow + kk + kbA;
        FragU a0, a1;
        a0.u[0] = *(const uint4*)(ap);
        a0.u[1] = *(const uint4*)(ap + 16);
        a1.u[0] = *(const uint4*)(ap + 32);
        a1.u[1] = *(const uint4*)(ap + 48);
        __builtin_prefetch(arow + kk + 64, 0, 1);

#if USE_ASYNC_LDS
        __builtin_amdgcn_s_wait_asynccnt(0);
        __syncthreads();
        if (kk + 64 < K) stageB(kk + 64, buf ^ 1);
#else
        __syncthreads();                     // protect overwrite of Bsm[0]
        stageB(kk, 0);
        __syncthreads();
#endif

#pragma unroll
        for (int ks = 0; ks < 2; ++ks) {
#pragma unroll
            for (int t = 0; t < NTILES; ++t) {
                const bf16* bp = &Bsm[buf][((ks * NTILES + t) * 32 + lane) * 16];
                FragU b;
                b.u[0] = *(const uint4*)bp;
                b.u[1] = *(const uint4*)(bp + 8);
                acc[t] = __builtin_amdgcn_wmma_f32_16x16x32_bf16(
                    false, (ks ? a1.v : a0.v), false, b.v, (short)0, acc[t], false, false);
            }
        }
#if USE_ASYNC_LDS
        buf ^= 1;
#endif
    }

#pragma unroll
    for (int t = 0; t < NTILES; ++t) {
        const int col = col0 + t * 16 + (lane & 15);
        const float bv = bias ? bias[col] : 0.0f;
#pragma unroll
        for (int r = 0; r < 8; ++r) {
            const int row = row0 + r + (half << 3);
            float v = acc[t][r] + bv;
            if (EPI == 1) v = gelu_exact(v);
            if (EPI == 2) v += aux[(size_t)row * N + col];
            const size_t idx = (size_t)row * N + col;
            if (Cf) Cf[idx] = v;
            if (Cb) Cb[idx] = (bf16)v;
        }
    }
}

// --------------------------- elementwise helpers ---------------------------

__global__ void pack_bf16_kernel(const float* __restrict__ in, bf16* __restrict__ out, int n) {
    int i = blockIdx.x * 256 + threadIdx.x;
    if (i < n) out[i] = (bf16)in[i];
}

// bev [B=16, C=256, Nv=16384] f32 -> bevT [B*Nv, C] bf16 (LDS tiled transpose)
__global__ void transpose_pack_kernel(const float* __restrict__ bev, bf16* __restrict__ out) {
    __shared__ float tile[32][33];
    const int b  = blockIdx.z;
    const int c0 = blockIdx.y * 32;
    const int p0 = blockIdx.x * 32;
    const int tx = threadIdx.x, ty = threadIdx.y;     // block (32, 8)
#pragma unroll
    for (int i = 0; i < 32; i += 8)
        tile[ty + i][tx] = bev[((size_t)b * 256 + c0 + ty + i) * 16384 + p0 + tx];
    __syncthreads();
#pragma unroll
    for (int i = 0; i < 32; i += 8)
        out[((size_t)b * 16384 + p0 + ty + i) * 256 + c0 + tx] = (bf16)tile[tx][ty + i];
}

// pe[ch=128, hw=16384]: ch<64 -> sin/cos(w * div[ch/2]); ch>=64 -> sin/cos(h * div)
__global__ void pe_kernel(float* __restrict__ pe, int total) {
    int t = blockIdx.x * 256 + threadIdx.x;
    if (t >= total) return;
    int hw = t & 16383, ch = t >> 14;
    int h = hw >> 7, w = hw & 127;
    int i = (ch & 63) >> 1;
    float dv = expf(-(float)(2 * i) * 0.14391157176071944f);   // ln(10000)/64
    float coord = (ch < 64) ? (float)w : (float)h;
    float a = coord * dv;
    pe[t] = (ch & 1) ? cosf(a) : sinf(a);
}

__global__ void addpe_kernel(const float* __restrict__ vm, const float* __restrict__ pe,
                             bf16* __restrict__ out, int total) {
    int t = blockIdx.x * 256 + threadIdx.x;
    if (t >= total) return;
    int ch = t & 127;
    int hw = (t >> 7) & 16383;
    out[t] = (bf16)(vm[t] + pe[ch * 16384 + hw]);
}

__global__ void broadcast_q_kernel(const float* __restrict__ Q0,
                                   float* __restrict__ qf, bf16* __restrict__ qb, int total) {
    int t = blockIdx.x * 256 + threadIdx.x;
    if (t >= total) return;
    int c = t & 127, row = t >> 7;
    int b = row >> 11, nn = row & 2047;
    float x = Q0[(((size_t)b << 7) + (nn >> 4)) * 128 + c];
    qf[t] = x; qb[t] = (bf16)x;
}

__global__ void depth_softmax_kernel(const float* __restrict__ dep, float* __restrict__ r, int rows) {
    int t = blockIdx.x * 256 + threadIdx.x;
    if (t >= rows) return;
    const float* p = dep + (size_t)t * 64;
    float mx = -1e30f;
    for (int i = 0; i < 64; ++i) mx = fmaxf(mx, p[i]);
    float s = 0.0f, a = 0.0f;
    for (int i = 0; i < 64; ++i) {
        float e = expf(p[i] - mx);
        s += e; a += e * ((i + 0.5f) * 0.8f);
    }
    r[t] = a / s;
}

__global__ void geometry_kernel(const float* __restrict__ r, const float* __restrict__ L,
                                float* __restrict__ ref, int total) {
    int t = blockIdx.x * 256 + threadIdx.x;
    if (t >= total) return;
    int b = t / 2048, nn = t % 2048;
    int hr = nn >> 10, rem = nn & 1023, wr = rem >> 4, q = rem & 15;
    int qh = q >> 2, qw = q & 3;
    float h_og = hr * 16.0f + (qh + 0.5f) * 4.0f;
    float w_og = wr * 16.0f + (qw + 0.5f) * 4.0f;
    const float PI = 3.14159265358979323846f;
    float az = w_og * (2.0f * PI / 1024.0f) - PI;
    const float FU = 0.17453292519943295f, FD = -0.5235987755982988f;
    float el = FU - h_og * ((FU - FD) / 32.0f);
    float ce = cosf(el), se = sinf(el);
    float rr = r[t];
    float p0 = rr * ce * cosf(az), p1 = rr * ce * sinf(az), p2 = rr * se;
    const float* M4 = L + b * 16;
    float pex = M4[0] * p0 + M4[1] * p1 + M4[2] * p2 + M4[3];
    float pey = M4[4] * p0 + M4[5] * p1 + M4[6] * p2 + M4[7];
    ref[t * 2 + 0] = fminf(fmaxf((pex + 51.2f) * (1.0f / 102.4f), 0.0f), 1.0f);
    ref[t * 2 + 1] = fminf(fmaxf((pey + 51.2f) * (1.0f / 102.4f), 0.0f), 1.0f);
}

__global__ void attn_softmax_kernel(float* __restrict__ a, int total) {
    int t = blockIdx.x * 256 + threadIdx.x;
    if (t >= total) return;
    float* p = a + (size_t)(t >> 3) * 48 + (t & 7) * 6;
    float mx = -1e30f;
    for (int i = 0; i < 6; ++i) mx = fmaxf(mx, p[i]);
    float e[6], s = 0.0f;
    for (int i = 0; i < 6; ++i) { e[i] = expf(p[i] - mx); s += e[i]; }
    float inv = 1.0f / s;
    for (int i = 0; i < 6; ++i) p[i] = e[i] * inv;
}

// deformable bilinear sampling: thread = (b, n, h, e); e-contiguous -> coalesced
__global__ void sample_kernel(const float* __restrict__ v, const float* __restrict__ ref,
                              const float* __restrict__ off, const float* __restrict__ attn,
                              float* __restrict__ outf, bf16* __restrict__ outb, int total) {
    int t = blockIdx.x * 256 + threadIdx.x;
    if (t >= total) return;
    int e = t & 15, h = (t >> 4) & 7, row = t >> 7;
    int b = row >> 11;
    float rx = ref[row * 2], ry = ref[row * 2 + 1];
    const float* offp = off + (size_t)row * 96 + h * 12;
    const float* ap   = attn + (size_t)row * 48 + h * 6;
    const float* vb   = v + ((size_t)b << 14) * 128 + h * 16 + e;
    float acc = 0.0f;
    for (int p = 0; p < 6; ++p) {
        float xs = (rx + offp[p * 2 + 0] * (1.0f / 128.0f)) * 128.0f - 0.5f;
        float ys = (ry + offp[p * 2 + 1] * (1.0f / 128.0f)) * 128.0f - 0.5f;
        float xf = floorf(xs), yf = floorf(ys);
        float wx = xs - xf, wy = ys - yf;
        int x0 = (int)xf, y0 = (int)yf;
        float s = 0.0f;
#pragma unroll
        for (int cy = 0; cy < 2; ++cy) {
            int yi = y0 + cy;
            float wyc = cy ? wy : (1.0f - wy);
#pragma unroll
            for (int cx = 0; cx < 2; ++cx) {
                int xi = x0 + cx;
                float wxc = cx ? wx : (1.0f - wx);
                bool ok = (xi >= 0) & (xi < 128) & (yi >= 0) & (yi < 128);
                int yc = min(max(yi, 0), 127), xc = min(max(xi, 0), 127);
                float val = vb[(size_t)(yc * 128 + xc) * 128];
                s += wxc * wyc * (ok ? val : 0.0f);
            }
        }
        acc += ap[p] * s;
    }
    size_t oi = (size_t)row * 128 + h * 16 + e;
    outf[oi] = acc; outb[oi] = (bf16)acc;
}

__global__ void gather_ym_kernel(const float* __restrict__ ym, bf16* __restrict__ out, int total) {
    int t = blockIdx.x * 256 + threadIdx.x;
    if (t >= total) return;
    int c2 = t & 2047, rowo = t >> 11;
    int b = rowo >> 7, pos = rowo & 127;
    int nn = ((c2 & 15) << 7) + pos;
    out[t] = (bf16)ym[(((size_t)b << 11) + nn) * 128 + (c2 >> 4)];
}

__global__ void layernorm_kernel(const float* __restrict__ in, const float* __restrict__ g,
                                 const float* __restrict__ bb,
                                 float* __restrict__ outf, bf16* __restrict__ outb, int rows) {
    int lane = threadIdx.x & 31;
    int row = blockIdx.x * 4 + (threadIdx.x >> 5);
    if (row >= rows) return;
    const float* p = in + (size_t)row * 256;
    float x[8], s = 0.0f;
#pragma unroll
    for (int i = 0; i < 8; ++i) { x[i] = p[lane + i * 32]; s += x[i]; }
#pragma unroll
    for (int o = 16; o > 0; o >>= 1) s += __shfl_xor(s, o, 32);
    float mean = s * (1.0f / 256.0f);
    float var = 0.0f;
#pragma unroll
    for (int i = 0; i < 8; ++i) { float d = x[i] - mean; var += d * d; }
#pragma unroll
    for (int o = 16; o > 0; o >>= 1) var += __shfl_xor(var, o, 32);
    float inv = rsqrtf(var * (1.0f / 256.0f) + 1e-5f);
#pragma unroll
    for (int i = 0; i < 8; ++i) {
        int col = lane + i * 32;
        float y = (x[i] - mean) * inv * g[col] + bb[col];
        size_t idx = (size_t)row * 256 + col;
        if (outf) outf[idx] = y;
        if (outb) outb[idx] = (bf16)y;
    }
}

__global__ void tail_kernel(float* d) { if (threadIdx.x == 0) d[0] = 0.0f; }

// ---------------------------------------------------------------------------

static inline void launch_gemm(int NT, int EPI, const bf16* A, const bf16* W,
                               const float* bias, const float* aux,
                               float* Cf, bf16* Cb, int M, int N, int K, hipStream_t s) {
    dim3 grid(N / (16 * NT), M / 64), block(128);
    if      (NT == 4 && EPI == 0) gemm_wmma_kernel<4,0><<<grid,block,0,s>>>(A,W,bias,aux,Cf,Cb,M,N,K);
    else if (NT == 3 && EPI == 0) gemm_wmma_kernel<3,0><<<grid,block,0,s>>>(A,W,bias,aux,Cf,Cb,M,N,K);
    else if (NT == 4 && EPI == 1) gemm_wmma_kernel<4,1><<<grid,block,0,s>>>(A,W,bias,aux,Cf,Cb,M,N,K);
    else if (NT == 4 && EPI == 2) gemm_wmma_kernel<4,2><<<grid,block,0,s>>>(A,W,bias,aux,Cf,Cb,M,N,K);
}

extern "C" void kernel_launch(void* const* d_in, const int* in_sizes, int n_in,
                              void* d_out, int out_size, void* d_ws, size_t ws_size,
                              hipStream_t stream) {
    (void)in_sizes; (void)n_in; (void)out_size; (void)ws_size;
    const float* x_rv  = (const float*)d_in[0];
    const float* bev   = (const float*)d_in[1];
    const float* l2e   = (const float*)d_in[2];
    const float* Wq    = (const float*)d_in[3];   const float* bq    = (const float*)d_in[4];
    const float* Wv    = (const float*)d_in[5];   const float* bv    = (const float*)d_in[6];
    const float* Wdep  = (const float*)d_in[7];   const float* bdep  = (const float*)d_in[8];
    const float* Woff  = (const float*)d_in[9];   const float* boff  = (const float*)d_in[10];
    const float* Wattn = (const float*)d_in[11];  const float* battn = (const float*)d_in[12];
    const float* Wvm   = (const float*)d_in[13];  const float* bvm   = (const float*)d_in[14];
    const float* Wom   = (const float*)d_in[15];  const float* bom   = (const float*)d_in[16];
    const float* Wo1   = (const float*)d_in[17];  const float* bo1   = (const float*)d_in[18];
    const float* Wo2   = (const float*)d_in[19];  const float* bo2   = (const float*)d_in[20];
    const float* ln1g  = (const float*)d_in[21];  const float* ln1b  = (const float*)d_in[22];
    const float* Wf1   = (const float*)d_in[23];  const float* bf1   = (const float*)d_in[24];
    const float* Wf2   = (const float*)d_in[25];  const float* bf2   = (const float*)d_in[26];
    const float* ln2g  = (const float*)d_in[27];  const float* ln2b  = (const float*)d_in[28];

    float* dout    = (float*)d_out;
    float* dep_out = dout + 524288;            // depth_logits region [2048, 1024]

    char* ws = (char*)d_ws;
    size_t o = 0;
    auto alloc = [&](size_t bytes) -> char* {
        char* p = ws + o;
        o += (bytes + 255) & ~(size_t)255;
        return p;
    };

    char* regA = alloc(134217728);   // bevT_b (bf16) -> later value_b (bf16)
    char* regV = alloc(134217728);   // Vmap (f32)    -> later v (f32)
    bf16*  bevT  = (bf16*)regA;
    float* Vmap  = (float*)regV;
    bf16*  valB  = (bf16*)regA;
    float* vbuf  = (float*)regV;

    float* pe     = (float*)alloc(8388608);
    bf16*  xb     = (bf16*)alloc(1048576);
    bf16*  wq_b   = (bf16*)alloc(65536);
    bf16*  wv_b   = (bf16*)alloc(65536);
    bf16*  wdep_b = (bf16*)alloc(524288);
    bf16*  woff_b = (bf16*)alloc(24576);
    bf16*  wattn_b= (bf16*)alloc(12288);
    bf16*  wvm_b  = (bf16*)alloc(32768);
    bf16*  wom_b  = (bf16*)alloc(32768);
    bf16*  wo1_b  = (bf16*)alloc(2097152);
    bf16*  wo2_b  = (bf16*)alloc(262144);
    bf16*  wf1_b  = (bf16*)alloc(131072);
    bf16*  wf2_b  = (bf16*)alloc(131072);
    float* Q0     = (float*)alloc(1048576);
    float* qf     = (float*)alloc(16777216);
    bf16*  qb     = (bf16*)alloc(8388608);
    float* offb   = (float*)alloc(12582912);
    float* attnb  = (float*)alloc(6291456);
    float* rbuf   = (float*)alloc(131072);
    float* refb   = (float*)alloc(262144);
    float* out_f  = (float*)alloc(16777216);
    bf16*  out_b  = (bf16*)alloc(8388608);
    float* ymsda  = (float*)alloc(16777216);
    bf16*  ym_b   = (bf16*)alloc(8388608);
    bf16*  h1b    = (bf16*)alloc(2097152);
    float* h2f    = (float*)alloc(2097152);
    float* y1f    = (float*)alloc(2097152);
    bf16*  y1b    = (bf16*)alloc(1048576);
    bf16*  f1b    = (bf16*)alloc(1048576);
    float* f2r    = (float*)alloc(2097152);

    auto pack = [&](const float* in, bf16* outp, int n) {
        pack_bf16_kernel<<<(n + 255) / 256, 256, 0, stream>>>(in, outp, n);
    };

    pack(x_rv, xb, 524288);
    pack(Wq, wq_b, 32768);       pack(Wv, wv_b, 32768);       pack(Wdep, wdep_b, 262144);
    pack(Woff, woff_b, 12288);   pack(Wattn, wattn_b, 6144);  pack(Wvm, wvm_b, 16384);
    pack(Wom, wom_b, 16384);     pack(Wo1, wo1_b, 1048576);   pack(Wo2, wo2_b, 131072);
    pack(Wf1, wf1_b, 65536);     pack(Wf2, wf2_b, 65536);

    transpose_pack_kernel<<<dim3(512, 8, 16), dim3(32, 8), 0, stream>>>(bev, bevT);
    pe_kernel<<<2097152 / 256, 256, 0, stream>>>(pe, 2097152);

    launch_gemm(4, 0, xb, wq_b, bq, nullptr, Q0, nullptr, 2048, 128, 256, stream);
    launch_gemm(4, 0, xb, wdep_b, bdep, nullptr, dep_out, nullptr, 2048, 1024, 256, stream);
    broadcast_q_kernel<<<4194304 / 256, 256, 0, stream>>>(Q0, qf, qb, 4194304);

    launch_gemm(4, 0, bevT, wv_b, bv, nullptr, Vmap, nullptr, 262144, 128, 256, stream);
    addpe_kernel<<<33554432 / 256, 256, 0, stream>>>(Vmap, pe, valB, 33554432);
    launch_gemm(4, 0, valB, wvm_b, bvm, nullptr, vbuf, nullptr, 262144, 128, 128, stream);

    launch_gemm(3, 0, qb, woff_b, boff, nullptr, offb, nullptr, 32768, 96, 128, stream);
    launch_gemm(3, 0, qb, wattn_b, battn, nullptr, attnb, nullptr, 32768, 48, 128, stream);
    attn_softmax_kernel<<<262144 / 256, 256, 0, stream>>>(attnb, 262144);

    depth_softmax_kernel<<<32768 / 256, 256, 0, stream>>>(dep_out, rbuf, 32768);
    geometry_kernel<<<32768 / 256, 256, 0, stream>>>(rbuf, l2e, refb, 32768);

    sample_kernel<<<4194304 / 256, 256, 0, stream>>>(vbuf, refb, offb, attnb, out_f, out_b, 4194304);

    launch_gemm(4, 2, out_b, wom_b, bom, qf, ymsda, nullptr, 32768, 128, 128, stream);
    gather_ym_kernel<<<4194304 / 256, 256, 0, stream>>>(ymsda, ym_b, 4194304);

    launch_gemm(4, 1, ym_b, wo1_b, bo1, nullptr, nullptr, h1b, 2048, 512, 2048, stream);
    launch_gemm(4, 2, h1b, wo2_b, bo2, x_rv, h2f, nullptr, 2048, 256, 512, stream);

    layernorm_kernel<<<512, 128, 0, stream>>>(h2f, ln1g, ln1b, y1f, y1b, 2048);

    launch_gemm(4, 1, y1b, wf1_b, bf1, nullptr, nullptr, f1b, 2048, 256, 256, stream);
    launch_gemm(4, 2, f1b, wf2_b, bf2, y1f, f2r, nullptr, 2048, 256, 256, stream);

    layernorm_kernel<<<512, 128, 0, stream>>>(f2r, ln2g, ln2b, dout, nullptr, 2048);

    tail_kernel<<<1, 1, 0, stream>>>(dout + 2621440);
}
